// LiquidLSTM_87110526697523
// MI455X (gfx1250) — matrix operand order repaired and checked by reading, not compiled, not run on hardware
//
#include <hip/hip_runtime.h>
#include <hip/hip_bf16.h>
#include <math.h>

#define BB 64
#define TT 512
#define II 512
#define HH 1024
#define DT_CONST 0.1f

typedef __attribute__((ext_vector_type(16))) __bf16 v16bf;
typedef __attribute__((ext_vector_type(8)))  __bf16 v8bf;
typedef __attribute__((ext_vector_type(8)))  float  v8f;

// ---------------------------------------------------------------------------
// WMMA fragment loaders (CDNA5 ISA 7.12.2 VGPR layouts, wave32)
// ---------------------------------------------------------------------------
__device__ __forceinline__ v16bf load_a16(const __bf16* __restrict__ base,
                                          int ldk, int row, int k0, int lane) {
  int m    = row + (lane & 15);
  int half = lane >> 4;
  const __bf16* p = base + (size_t)m * ldk + k0 + half * 8;
  v8bf a0 = *reinterpret_cast<const v8bf*>(p);
  v8bf a1 = *reinterpret_cast<const v8bf*>(p + 16);
  v16bf r;
#pragma unroll
  for (int i = 0; i < 8; ++i) { r[i] = a0[i]; r[8 + i] = a1[i]; }
  return r;
}

__device__ __forceinline__ v16bf load_b16(const __bf16* __restrict__ base,
                                          int ldk, int col, int k0, int lane) {
  int n    = col + (lane & 15);
  int half = lane >> 4;
  const __bf16* p = base + (size_t)n * ldk + k0 + half * 8;
  v8bf b0 = *reinterpret_cast<const v8bf*>(p);
  v8bf b1 = *reinterpret_cast<const v8bf*>(p + 16);
  v16bf r;
#pragma unroll
  for (int i = 0; i < 8; ++i) { r[i] = b0[i]; r[8 + i] = b1[i]; }
  return r;
}

__device__ __forceinline__ v8f wmma_bf16(v16bf a, v16bf b, v8f c) {
  return __builtin_amdgcn_wmma_f32_16x16x32_bf16(false, a, false, b,
                                                 (short)0, c, false, false);
}

// ---------------------------------------------------------------------------
// Fixed-length K-segment GEMM: KLEN is a compile-time constant so the loop
// fully unrolls into straight-line code -- no loop-carried registers, no
// back-edge register shuffling, and the scheduler can hoist load clauses
// arbitrarily far ahead of the consuming WMMAs. `abase`/`bbase` are
// pre-offset to this wave's k_begin; all k addressing folds into immediates.
// ---------------------------------------------------------------------------
template <int NGATE, int KLEN>
__device__ __forceinline__ void gemm_fixed(
    const __bf16* __restrict__ abase, int aldk, int ib,
    const __bf16* __restrict__ bbase, size_t bgstride, int bldk, int jb,
    int lane, v8f* acc)
{
#pragma unroll
  for (int kk = 0; kk < KLEN; kk += 32) {
    v16bf a = load_a16(abase, aldk, ib, kk, lane);
#pragma unroll
    for (int g = 0; g < NGATE; ++g) {
      v16bf b = load_b16(bbase + (size_t)g * bgstride, bldk, jb, kk, lane);
      acc[g] = wmma_bf16(a, b, acc[g]);
    }
  }
}

// ---------------------------------------------------------------------------
// One LSTM step. Grid: 64 blocks x 512 threads = 1024 waves.
// Block owns 4 (batch16 x hidden16) tiles; each tile computed by 4 waves.
// Wave ks handles x[ks*128,+128) and h[ks*256,+256)  (compile-time lengths),
// partials reduced through LDS; ks0 wave adds bias, reduces, runs gate math.
// ---------------------------------------------------------------------------
__global__ __launch_bounds__(512, 1) void lstm_step_kernel(
    const __bf16* __restrict__ xbf,   // [B][T][I] bf16
    const __bf16* __restrict__ wih,   // [4H][I]  bf16
    const __bf16* __restrict__ whh,   // [4H][H]  bf16
    const float*  __restrict__ bias4, // [4H]     b_ih + b_hh
    const __bf16* __restrict__ h_in,  // [B][H]   bf16 (= ys[t-1] or zeros)
    float*        __restrict__ c,     // [B][H]   f32, in-place
    __bf16*       __restrict__ h_out, // [B][H]   bf16 (= ys[t])
    int t)
{
  __shared__ __align__(32) float red[3 * 4 * 4 * 256]; // [ks-1][tl][gate][lane*8]

  int lane  = threadIdx.x & 31;
  // wave-uniform values forced into SGPRs -> SALU control flow, EXEC untouched
  int wave  = __builtin_amdgcn_readfirstlane(threadIdx.x >> 5);
  int tl    = wave & 3;                // tile within block
  int ks    = wave >> 2;               // K-split 0..3
  int tile  = __builtin_amdgcn_readfirstlane(blockIdx.x) * 4 + tl;
  int jb    = (tile & 63) * 16;
  int ib    = (tile >> 6) * 16;
  int colL  = lane & 15;
  int halfL = lane >> 4;

  v8f acc[4];
  if (ks == 0) {
#pragma unroll
    for (int g = 0; g < 4; ++g) {
      float bv = bias4[g * HH + jb + colL];
#pragma unroll
      for (int r = 0; r < 8; ++r) acc[g][r] = bv;
    }
  } else {
#pragma unroll
    for (int g = 0; g < 4; ++g)
#pragma unroll
      for (int r = 0; r < 8; ++r) acc[g][r] = 0.0f;
  }

  // x_t contribution: this wave covers x columns [ks*128, ks*128+128)
  {
    int k0 = ks * 128;
    gemm_fixed<4, 128>(xbf + (size_t)t * II + k0, TT * II, ib,
                       wih + k0, (size_t)HH * II, II, jb, lane, acc);
  }
  // h_{t-1} contribution: this wave covers h columns [ks*256, ks*256+256)
  {
    int k0 = ks * 256;
    gemm_fixed<4, 256>(h_in + k0, HH, ib,
                       whh + k0, (size_t)HH * HH, HH, jb, lane, acc);
  }

  if (ks != 0) {
#pragma unroll
    for (int g = 0; g < 4; ++g)
      *reinterpret_cast<v8f*>(
          &red[(((ks - 1) * 4 + tl) * 4 + g) * 256 + lane * 8]) = acc[g];
  }
  __syncthreads();

  if (ks == 0) {
#pragma unroll
    for (int g = 0; g < 4; ++g)
      for (int s = 0; s < 3; ++s) {
        v8f pp = *reinterpret_cast<const v8f*>(
            &red[((s * 4 + tl) * 4 + g) * 256 + lane * 8]);
#pragma unroll
        for (int r = 0; r < 8; ++r) acc[g][r] += pp[r];
      }

    int col = jb + colL;
#pragma unroll
    for (int r = 0; r < 8; ++r) {
      int m = ib + halfL * 8 + r;
      size_t idx = (size_t)m * HH + col;
      float gi = acc[0][r], gf = acc[1][r], gg = acc[2][r], go = acc[3][r];
      float si = 1.0f / (1.0f + __expf(-gi));
      float sf = 1.0f / (1.0f + __expf(-gf));
      float so = 1.0f / (1.0f + __expf(-go));
      float cn = sf * c[idx] + si * tanhf(gg);
      c[idx] = cn;
      h_out[idx] = (__bf16)(so * tanhf(cn));
    }
  }
}

// ---------------------------------------------------------------------------
// One liquid step: h = h + leak * (tanh(ys_t@W_in^T + h@W_rec^T + bias) - h)
// Wave ks covers ys[ks*256,+256) and h[ks*256,+256) (compile-time lengths).
// ---------------------------------------------------------------------------
__global__ __launch_bounds__(512, 1) void liquid_step_kernel(
    const __bf16* __restrict__ ys_t,  // [B][H] bf16
    const __bf16* __restrict__ win,   // [H][H] bf16
    const __bf16* __restrict__ wrec,  // [H][H] bf16
    const float*  __restrict__ biasl, // [H]    b_in + b_rec
    const float*  __restrict__ leak,  // [H]    DT / clip(tau)
    const __bf16* __restrict__ h_in,  // [B][H] bf16 (ping)
    float*        __restrict__ h,     // [B][H] f32 master state, in-place
    __bf16*       __restrict__ h_out) // [B][H] bf16 (pong)
{
  __shared__ __align__(32) float red[3 * 4 * 256]; // [ks-1][tl][lane*8]

  int lane  = threadIdx.x & 31;
  int wave  = __builtin_amdgcn_readfirstlane(threadIdx.x >> 5);
  int tl    = wave & 3;
  int ks    = wave >> 2;
  int tile  = __builtin_amdgcn_readfirstlane(blockIdx.x) * 4 + tl;
  int jb    = (tile & 63) * 16;
  int ib    = (tile >> 6) * 16;
  int colL  = lane & 15;
  int halfL = lane >> 4;

  v8f acc[1];
  {
    float bv = (ks == 0) ? biasl[jb + colL] : 0.0f;
#pragma unroll
    for (int r = 0; r < 8; ++r) acc[0][r] = bv;
  }

  int k0 = ks * 256;
  gemm_fixed<1, 256>(ys_t + k0, HH, ib, win + k0, 0, HH, jb, lane, acc);
  gemm_fixed<1, 256>(h_in + k0, HH, ib, wrec + k0, 0, HH, jb, lane, acc);

  if (ks != 0) {
    *reinterpret_cast<v8f*>(&red[((ks - 1) * 4 + tl) * 256 + lane * 8]) = acc[0];
  }
  __syncthreads();

  if (ks == 0) {
    for (int s = 0; s < 3; ++s) {
      v8f pp = *reinterpret_cast<const v8f*>(&red[(s * 4 + tl) * 256 + lane * 8]);
#pragma unroll
      for (int r = 0; r < 8; ++r) acc[0][r] += pp[r];
    }
    int col = jb + colL;
    float lk = leak[col];
#pragma unroll
    for (int r = 0; r < 8; ++r) {
      int m = ib + halfL * 8 + r;
      size_t idx = (size_t)m * HH + col;
      float ho = h[idx];
      float hn = ho + lk * (tanhf(acc[0][r]) - ho);
      h[idx] = hn;
      h_out[idx] = (__bf16)hn;
    }
  }
}

// ------------------------------- setup helpers -----------------------------
__global__ void cvt_bf16_kernel(const float* __restrict__ in,
                                __bf16* __restrict__ out, int n) {
  for (int i = blockIdx.x * blockDim.x + threadIdx.x; i < n;
       i += gridDim.x * blockDim.x)
    out[i] = (__bf16)in[i];
}
__global__ void add2_kernel(const float* __restrict__ a,
                            const float* __restrict__ b,
                            float* __restrict__ o, int n) {
  for (int i = blockIdx.x * blockDim.x + threadIdx.x; i < n;
       i += gridDim.x * blockDim.x)
    o[i] = a[i] + b[i];
}
__global__ void leak_kernel(const float* __restrict__ tau,
                            float* __restrict__ o, int n) {
  for (int i = blockIdx.x * blockDim.x + threadIdx.x; i < n;
       i += gridDim.x * blockDim.x) {
    float tc = fminf(fmaxf(tau[i], 0.1f), 5.0f);
    o[i] = DT_CONST / tc;
  }
}
__global__ void zero_f_kernel(float* __restrict__ p, int n) {
  for (int i = blockIdx.x * blockDim.x + threadIdx.x; i < n;
       i += gridDim.x * blockDim.x)
    p[i] = 0.0f;
}
__global__ void zero_bf_kernel(unsigned short* __restrict__ p, int n) {
  for (int i = blockIdx.x * blockDim.x + threadIdx.x; i < n;
       i += gridDim.x * blockDim.x)
    p[i] = 0;
}
__global__ void copy_f_kernel(const float* __restrict__ in,
                              float* __restrict__ out, int n) {
  for (int i = blockIdx.x * blockDim.x + threadIdx.x; i < n;
       i += gridDim.x * blockDim.x)
    out[i] = in[i];
}

// ---------------------------------------------------------------------------
extern "C" void kernel_launch(void* const* d_in, const int* in_sizes, int n_in,
                              void* d_out, int out_size, void* d_ws, size_t ws_size,
                              hipStream_t stream) {
  (void)in_sizes; (void)n_in; (void)out_size; (void)ws_size;

  const float* x     = (const float*)d_in[0];
  const float* W_ih  = (const float*)d_in[1];
  const float* W_hh  = (const float*)d_in[2];
  const float* b_ih  = (const float*)d_in[3];
  const float* b_hh  = (const float*)d_in[4];
  const float* W_in  = (const float*)d_in[5];
  const float* b_in  = (const float*)d_in[6];
  const float* W_rec = (const float*)d_in[7];
  const float* b_rec = (const float*)d_in[8];
  const float* tau   = (const float*)d_in[9];

  char* p = (char*)d_ws;
  auto alloc = [&](size_t bytes) -> char* {
    char* r = p;
    p += (bytes + 255) & ~(size_t)255;
    return r;
  };

  const size_t BH = (size_t)BB * HH;
  __bf16* xbf    = (__bf16*)alloc((size_t)BB * TT * II * sizeof(__bf16));
  __bf16* wihb   = (__bf16*)alloc((size_t)4 * HH * II * sizeof(__bf16));
  __bf16* whhb   = (__bf16*)alloc((size_t)4 * HH * HH * sizeof(__bf16));
  __bf16* winb   = (__bf16*)alloc((size_t)HH * HH * sizeof(__bf16));
  __bf16* wrecb  = (__bf16*)alloc((size_t)HH * HH * sizeof(__bf16));
  __bf16* ys     = (__bf16*)alloc((size_t)TT * BH * sizeof(__bf16));   // 64 MB
  __bf16* zeroh  = (__bf16*)alloc(BH * sizeof(__bf16));
  __bf16* hliqbf = (__bf16*)alloc(2 * BH * sizeof(__bf16));
  float*  bias4  = (float*)alloc((size_t)4 * HH * sizeof(float));
  float*  biasl  = (float*)alloc((size_t)HH * sizeof(float));
  float*  leak   = (float*)alloc((size_t)HH * sizeof(float));
  float*  cbuf   = (float*)alloc(BH * sizeof(float));
  float*  hliq   = (float*)alloc(BH * sizeof(float));

  auto blocks = [](int n) { return (n + 255) / 256; };

  // --- setup: bf16 conversions, fused biases, leak, zero states ---
  int nx = BB * TT * II;
  cvt_bf16_kernel<<<blocks(nx), 256, 0, stream>>>(x, xbf, nx);
  cvt_bf16_kernel<<<blocks(4 * HH * II), 256, 0, stream>>>(W_ih, wihb, 4 * HH * II);
  cvt_bf16_kernel<<<blocks(4 * HH * HH), 256, 0, stream>>>(W_hh, whhb, 4 * HH * HH);
  cvt_bf16_kernel<<<blocks(HH * HH), 256, 0, stream>>>(W_in, winb, HH * HH);
  cvt_bf16_kernel<<<blocks(HH * HH), 256, 0, stream>>>(W_rec, wrecb, HH * HH);
  add2_kernel<<<blocks(4 * HH), 256, 0, stream>>>(b_ih, b_hh, bias4, 4 * HH);
  add2_kernel<<<blocks(HH), 256, 0, stream>>>(b_in, b_rec, biasl, HH);
  leak_kernel<<<blocks(HH), 256, 0, stream>>>(tau, leak, HH);
  zero_bf_kernel<<<blocks((int)BH), 256, 0, stream>>>((unsigned short*)zeroh, (int)BH);
  zero_bf_kernel<<<blocks((int)(2 * BH)), 256, 0, stream>>>((unsigned short*)hliqbf, (int)(2 * BH));
  zero_f_kernel<<<blocks((int)BH), 256, 0, stream>>>(cbuf, (int)BH);
  zero_f_kernel<<<blocks((int)BH), 256, 0, stream>>>(hliq, (int)BH);

  // --- LSTM scan: ys[t] doubles as h_t (bf16) for the next step's GEMM ---
  for (int t = 0; t < TT; ++t) {
    const __bf16* h_prev = (t == 0) ? zeroh : (ys + (size_t)(t - 1) * BH);
    lstm_step_kernel<<<64, 512, 0, stream>>>(xbf, wihb, whhb, bias4, h_prev,
                                             cbuf, ys + (size_t)t * BH, t);
  }

  // --- liquid scan: f32 master state + bf16 ping-pong GEMM operand ---
  for (int t = 0; t < TT; ++t) {
    const __bf16* hin = hliqbf + (size_t)(t & 1) * BH;
    __bf16* hout      = hliqbf + (size_t)((t + 1) & 1) * BH;
    liquid_step_kernel<<<64, 512, 0, stream>>>(ys + (size_t)t * BH, winb, wrecb,
                                               biasl, leak, hin, hliq, hout);
  }

  copy_f_kernel<<<blocks((int)BH), 256, 0, stream>>>(hliq, (float*)d_out, (int)BH);
}
